// MultiEmbTopModule_11819749998688
// MI455X (gfx1250) — matrix operand order repaired and checked by reading, not compiled, not run on hardware
//
#include <hip/hip_runtime.h>

typedef __attribute__((ext_vector_type(16))) __bf16 v16bf;
typedef __attribute__((ext_vector_type(8)))  float  v8f;
typedef __attribute__((ext_vector_type(4)))  unsigned int u32x4;
typedef __attribute__((ext_vector_type(8)))  int i32x8;
typedef __attribute__((ext_vector_type(4)))  int i32x4;

#define BB   64      // batch
#define LP   64      // padded L (49 -> 64)
#define CTOT 3840    // total concat channels
#define NOUT 512     // output features
#define CH   8       // K-blocks (of 32) staged per TDM chunk: 8*4KB = 32KB
#define CHELEM (CH * 2048)   // bf16 elements per chunk (A panel)
#define NEG_INF (-3.402823466e38f)

// TDM availability / arity detection (device pass only; host pass -> fallback)
#if defined(__AMDGCN__) && __has_builtin(__builtin_amdgcn_tensor_load_to_lds) && \
    __has_builtin(__builtin_amdgcn_s_wait_tensorcnt)
#define USE_TDM 1
#else
#define USE_TDM 0
#endif
#if __has_include(<hip/amd_detail/amd_gfx1250_TDM.h>)
#define TDM_ARITY 6
#else
#define TDM_ARITY 5
#endif

#if USE_TDM
// 1-D TDM copy: nelem bf16 elements from global -> LDS (D# per ISA ch.8).
// group0: count=1 | lds_addr | global_addr | type=2
// group1: data_size=2B, tensor_dim0=nelem, tensor_dim1=1, tile_dim0=nelem,
//         tile_dim1=1, tensor_dim0_stride=nelem
__device__ __forceinline__ void tdm_load_1d(const void* gptr, unsigned lds_off,
                                            unsigned nelem) {
  unsigned long long ga = (unsigned long long)(uintptr_t)gptr;
  u32x4 g0;
  g0.x = 1u;                                   // count=1, user descriptor
  g0.y = (unsigned)__builtin_amdgcn_readfirstlane(lds_off);
  g0.z = (unsigned)__builtin_amdgcn_readfirstlane((unsigned)ga);
  g0.w = (unsigned)__builtin_amdgcn_readfirstlane(
           (unsigned)((ga >> 32) & 0x1FFFFFFu) | (2u << 30));
  i32x8 g1;
  g1[0] = (int)(1u << 16);                     // data_size = 1 -> 2 bytes
  g1[1] = (int)((nelem & 0xFFFFu) << 16);      // tensor_dim0[15:0]
  g1[2] = (int)(((nelem >> 16) & 0xFFFFu) | (1u << 16)); // dim0[31:16], dim1=1
  g1[3] = (int)((nelem & 0xFFFFu) << 16);      // tile_dim0
  g1[4] = 1;                                   // tile_dim1 = 1
  g1[5] = (int)nelem;                          // tensor_dim0_stride[31:0]
  g1[6] = 0;
  g1[7] = 0;
  i32x4 z4 = {0, 0, 0, 0};
#if TDM_ARITY == 6
  i32x8 z8 = {0, 0, 0, 0, 0, 0, 0, 0};
  __builtin_amdgcn_tensor_load_to_lds(g0, g1, z4, z4, z8, 0);
#else
  __builtin_amdgcn_tensor_load_to_lds(g0, g1, z4, z4, 0);
#endif
}
#endif

// ---------------------------------------------------------------------------
// 1) Max-pool act (B,C,H,H) -> ajf fp32 (B,C,49), non-overlapping kxk windows
// ---------------------------------------------------------------------------
__global__ void pool_kernel(const float* __restrict__ act, float* __restrict__ ajf,
                            int C, int H, int k) {
  int t = blockIdx.x * blockDim.x + threadIdx.x;
  int total = BB * C * 49;
  if (t >= total) return;
  int l = t % 49;
  int c = (t / 49) % C;
  int b = t / (49 * C);
  int ph = l / 7, pw = l % 7;
  const float* base = act + ((size_t)(b * C + c) * H + ph * k) * H + pw * k;
  float m = NEG_INF;
  for (int dh = 0; dh < k; ++dh)
    for (int dw = 0; dw < k; ++dw)
      m = fmaxf(m, base[dh * H + dw]);
  ajf[(size_t)(b * C + c) * 49 + l] = m;
}

// ---------------------------------------------------------------------------
// 2) Pack pooled data into BF16 WMMA fragments (per-lane register layout).
// ---------------------------------------------------------------------------
__global__ void frag_kernel(const float* __restrict__ ajf,
                            __bf16* __restrict__ Afrag, __bf16* __restrict__ Bfrag,
                            int C) {
  size_t t = (size_t)blockIdx.x * blockDim.x + threadIdx.x;
  size_t total = (size_t)BB * LP * C;
  if (t >= total) return;
  int l = (int)(t & 63);
  int c = (int)((t >> 6) % C);
  int i = (int)(t / ((size_t)C << 6));
  float v = (l < 49) ? ajf[((size_t)i * C + c) * 49 + l] : 0.0f;
  __bf16 bv = (__bf16)v;
  int kb = c >> 5, kk = c & 31;
  int mt = l >> 4, mr = l & 15;
  // A fragment: 16x32 tile, M=l, K=c
  int laneA = mr + ((kk >> 4) << 4);
  int eA = kk & 15;
  size_t idxA = ((((size_t)i * (C >> 5) + kb) * 4 + mt) << 9) + laneA * 16 + eA;
  Afrag[idxA] = bv;
  // B fragment: 32x16 tile, K=c, N=l
  int laneB = mr + ((kk >> 4) << 4);
  int eB = kk & 15;
  size_t idxB = ((((size_t)i * (C >> 5) + kb) * 4 + mt) << 9) + laneB * 16 + eB;
  Bfrag[idxB] = bv;
}

// ---------------------------------------------------------------------------
// 3) Fused per-pair kernel. A panel staged to LDS in 32KB chunks via TDM
//    (double-buffered), B fragments register-double-buffered from global.
//    Then softmax(49x49) -> row-mean -> emb GEMV -> BF16 A-frag of final GEMM.
// ---------------------------------------------------------------------------
__global__ __launch_bounds__(128) void attn_kernel(
    const float* __restrict__ ajf, const __bf16* __restrict__ Afrag,
    const __bf16* __restrict__ Bfrag, int C, int colofs,
    __bf16* __restrict__ Afinal) {
  // A-stage double buffer (64KB) overlaid with the softmax tile (disjoint phases)
  __shared__ union {
    __bf16 a[2][CHELEM];
    struct { float tile[64][65]; float wv[64]; } s;
  } sm;

  int p = blockIdx.x;
  int i = p >> 6, j = p & 63;
  int wave = threadIdx.x >> 5, lane = threadIdx.x & 31;
  int KB = C >> 5;
  int nChunks = KB / CH;

  const __bf16* Apanel = Afrag + (((size_t)i * KB) << 11);
  const v16bf* Bbase = (const v16bf*)(Bfrag + (((size_t)j * KB) << 11));

  v8f acc[4] = {};

#if USE_TDM
  if (wave == 0)
    tdm_load_1d(Apanel, (unsigned)(uintptr_t)&sm.a[0][0], CHELEM);
#endif

  for (int ch = 0; ch < nChunks; ++ch) {
    int buf = ch & 1;
#if USE_TDM
    if (wave == 0) __builtin_amdgcn_s_wait_tensorcnt(0);  // chunk ch landed
    __syncthreads();                                      // visible to all waves
    if (wave == 0 && ch + 1 < nChunks)
      tdm_load_1d(Apanel + (size_t)(ch + 1) * CHELEM,
                  (unsigned)(uintptr_t)&sm.a[(ch + 1) & 1][0], CHELEM);
#else
    __syncthreads();
    {  // cooperative fallback copy
      const v16bf* src = (const v16bf*)(Apanel + (size_t)ch * CHELEM);
      v16bf* dst = (v16bf*)&sm.a[buf][0];
      for (int q = threadIdx.x; q < CHELEM / 16; q += 128) dst[q] = src[q];
    }
    __syncthreads();
#endif
    const v16bf* As = (const v16bf*)&sm.a[buf][0];
    int kb0 = ch * CH;
    v16bf bCur = Bbase[((size_t)kb0 * 4 + wave) * 32 + lane];
#pragma unroll
    for (int u = 0; u < CH; ++u) {
      v16bf bNxt = bCur;
      if (u + 1 < CH)
        bNxt = Bbase[((size_t)(kb0 + u + 1) * 4 + wave) * 32 + lane];
#pragma unroll
      for (int mt = 0; mt < 4; ++mt) {
        v16bf af = As[(u * 4 + mt) * 32 + lane];
        acc[mt] = __builtin_amdgcn_wmma_f32_16x16x32_bf16(
            false, af, false, bCur, (short)0, acc[mt], false, false);
      }
      bCur = bNxt;
    }
  }
  __syncthreads();  // A-stage buffers dead; safe to overlay with tile

  // C/D layout: VGPR r, lanes 0-15 -> M=r, lanes 16-31 -> M=r+8; N=lane%16
  {
    int col = wave * 16 + (lane & 15);
    int rbase0 = ((lane >> 4) << 3);
#pragma unroll
    for (int mt = 0; mt < 4; ++mt)
#pragma unroll
      for (int r = 0; r < 8; ++r)
        sm.s.tile[mt * 16 + rbase0 + r][col] = acc[mt][r];
  }
  __syncthreads();

  int t = threadIdx.x;
  if (t < 49) {  // softmax over m for row l=t
    float mx = NEG_INF;
    for (int m = 0; m < 49; ++m) mx = fmaxf(mx, sm.s.tile[t][m]);
    float s = 0.f;
    for (int m = 0; m < 49; ++m) {
      float e = __expf(sm.s.tile[t][m] - mx);
      sm.s.tile[t][m] = e;
      s += e;
    }
    float inv = 1.0f / s;
    for (int m = 0; m < 49; ++m) sm.s.tile[t][m] *= inv;
  }
  __syncthreads();
  if (t < 49) {  // column mean over l
    float s = 0.f;
    for (int l = 0; l < 49; ++l) s += sm.s.tile[l][t];
    sm.s.wv[t] = s * (1.0f / 49.0f);
  }
  __syncthreads();

  // emb[c] = sum_m a_j[c,m]*wv[m] + mean_m a_j[c,m]; store as final-GEMM A frag
  int Mt = p >> 4, mr = p & 15;
  for (int c = t; c < C; c += 128) {
    const float* row = ajf + ((size_t)j * C + c) * 49;
    float s = 0.f, mean = 0.f;
    for (int m = 0; m < 49; ++m) {
      float v = row[m];
      s += v * sm.s.wv[m];
      mean += v;
    }
    float emb = s + mean * (1.0f / 49.0f);
    int k = colofs + c;
    int kb = k >> 5, kk = k & 31;
    int laneF = mr + ((kk >> 4) << 4);
    int eF = kk & 15;
    size_t idx = (((size_t)Mt * (CTOT >> 5) + kb) << 9) + laneF * 16 + eF;
    Afinal[idx] = (__bf16)emb;
  }
}

// ---------------------------------------------------------------------------
// 4) Pack projection weight W (3840,512) fp32 into BF16 B-fragments.
// ---------------------------------------------------------------------------
__global__ void wfrag_kernel(const float* __restrict__ W, __bf16* __restrict__ WB) {
  int t = blockIdx.x * blockDim.x + threadIdx.x;
  if (t >= CTOT * NOUT) return;
  int n = t % NOUT, k = t / NOUT;
  int kb = k >> 5, kk = k & 31;
  int nt = n >> 4;
  int lane = (n & 15) + ((kk >> 4) << 4);
  int e = kk & 15;
  size_t idx = (((size_t)kb * (NOUT >> 4) + nt) << 9) + lane * 16 + e;
  WB[idx] = (__bf16)W[t];
}

// ---------------------------------------------------------------------------
// 5) Final projection GEMM (4096x512, K=3840), register double-buffered WMMA.
// ---------------------------------------------------------------------------
__global__ __launch_bounds__(128) void fgemm_kernel(
    const __bf16* __restrict__ Afinal, const __bf16* __restrict__ WB,
    float* __restrict__ out) {
  int bm = blockIdx.x;  // 64 x 64-row strips
  int bn = blockIdx.y;  // 8 x 64-col strips
  int wave = threadIdx.x >> 5, lane = threadIdx.x & 31;
  const int KBF = CTOT >> 5;  // 120
  v8f acc[4] = {};
  const v16bf* Ab = (const v16bf*)Afinal;
  const v16bf* Bb = (const v16bf*)WB;

  v16bf aCur[4], aNxt[4], bCur, bNxt;
  bCur = Bb[((size_t)0 * 32 + bn * 4 + wave) * 32 + lane];
#pragma unroll
  for (int mt = 0; mt < 4; ++mt)
    aCur[mt] = Ab[((size_t)(bm * 4 + mt) * KBF + 0) * 32 + lane];

  for (int kb = 0; kb < KBF; ++kb) {
    bNxt = bCur;
#pragma unroll
    for (int mt = 0; mt < 4; ++mt) aNxt[mt] = aCur[mt];
    if (kb + 1 < KBF) {
      bNxt = Bb[((size_t)(kb + 1) * 32 + bn * 4 + wave) * 32 + lane];
#pragma unroll
      for (int mt = 0; mt < 4; ++mt)
        aNxt[mt] = Ab[((size_t)(bm * 4 + mt) * KBF + kb + 1) * 32 + lane];
    }
#pragma unroll
    for (int mt = 0; mt < 4; ++mt)
      acc[mt] = __builtin_amdgcn_wmma_f32_16x16x32_bf16(
          false, aCur[mt], false, bCur, (short)0, acc[mt], false, false);
#pragma unroll
    for (int mt = 0; mt < 4; ++mt) aCur[mt] = aNxt[mt];
    bCur = bNxt;
  }
#pragma unroll
  for (int mt = 0; mt < 4; ++mt) {
    int row = bm * 64 + mt * 16 + ((lane >> 4) << 3);
    int col = bn * 64 + wave * 16 + (lane & 15);
#pragma unroll
    for (int r = 0; r < 8; ++r)
      out[(size_t)(row + r) * NOUT + col] = acc[mt][r];
  }
}

// ---------------------------------------------------------------------------
// 6) Bias add + L2 row-normalize -> d_out
// ---------------------------------------------------------------------------
__global__ __launch_bounds__(128) void norm_kernel(const float* __restrict__ g,
                                                   const float* __restrict__ bias,
                                                   float* __restrict__ out) {
  __shared__ float red[128];
  int row = blockIdx.x;
  int t = threadIdx.x;
  float vals[4];
  float ss = 0.f;
#pragma unroll
  for (int q = 0; q < 4; ++q) {
    int c = t + q * 128;
    float v = g[(size_t)row * NOUT + c] + bias[c];
    vals[q] = v;
    ss += v * v;
  }
  red[t] = ss;
  __syncthreads();
  for (int s = 64; s > 0; s >>= 1) {
    if (t < s) red[t] += red[t + s];
    __syncthreads();
  }
  float inv = rsqrtf(red[0]);
#pragma unroll
  for (int q = 0; q < 4; ++q)
    out[(size_t)row * NOUT + t + q * 128] = vals[q] * inv;
}

// ---------------------------------------------------------------------------
extern "C" void kernel_launch(void* const* d_in, const int* in_sizes, int n_in,
                              void* d_out, int out_size, void* d_ws, size_t ws_size,
                              hipStream_t stream) {
  const float* acts[4] = {(const float*)d_in[0], (const float*)d_in[1],
                          (const float*)d_in[2], (const float*)d_in[3]};
  const float* W = (const float*)d_in[4];
  const float* bias = (const float*)d_in[5];
  float* out = (float*)d_out;

  char* ws = (char*)d_ws;
  size_t o = 0;
  auto alloc = [&](size_t bytes) -> void* {
    void* p = ws + o;
    o = (o + bytes + 255) & ~(size_t)255;
    return p;
  };
  float*  ajf    = (float*) alloc((size_t)BB * 2048 * 49 * sizeof(float));
  __bf16* Afrag  = (__bf16*)alloc((size_t)BB * 2048 * LP * sizeof(__bf16));
  __bf16* Bfrag  = (__bf16*)alloc((size_t)BB * 2048 * LP * sizeof(__bf16));
  __bf16* Afinal = (__bf16*)alloc((size_t)BB * BB * CTOT * sizeof(__bf16));
  __bf16* WB     = (__bf16*)alloc((size_t)CTOT * NOUT * sizeof(__bf16));
  float*  gout   = (float*) alloc((size_t)BB * BB * NOUT * sizeof(float));

  const int Cs[4] = {256, 512, 1024, 2048};
  const int Hs[4] = {56, 28, 14, 7};
  const int ks[4] = {8, 4, 2, 1};
  int colofs = 0;
  for (int s = 0; s < 4; ++s) {
    int C = Cs[s];
    int tp = BB * C * 49;
    pool_kernel<<<(tp + 255) / 256, 256, 0, stream>>>(acts[s], ajf, C, Hs[s], ks[s]);
    size_t tf = (size_t)BB * LP * C;
    frag_kernel<<<(unsigned)((tf + 255) / 256), 256, 0, stream>>>(ajf, Afrag, Bfrag, C);
    attn_kernel<<<BB * BB, 128, 0, stream>>>(ajf, Afrag, Bfrag, C, colofs, Afinal);
    colofs += C;
  }
  int tw = CTOT * NOUT;
  wfrag_kernel<<<(tw + 255) / 256, 256, 0, stream>>>(W, WB);
  fgemm_kernel<<<dim3(BB, NOUT / 64), 128, 0, stream>>>(Afinal, WB, gout);
  norm_kernel<<<BB * BB, 128, 0, stream>>>(gout, bias, out);
}
